// GeometricAttention_7576322310236
// MI455X (gfx1250) — matrix operand order repaired and checked by reading, not compiled, not run on hardware
//
#include <hip/hip_runtime.h>

typedef __attribute__((ext_vector_type(16))) _Float16 v16h;
typedef __attribute__((ext_vector_type(8)))  _Float16 v8h;
typedef __attribute__((ext_vector_type(8)))  float    v8f;

// Problem constants (match reference setup_inputs)
constexpr int B_   = 8;
constexpr int NQ   = 2048;
constexpr int NK   = 2048;
constexpr int D_   = 64;
constexpr int VDIM = 128;

constexpr float EPS_ = 1e-7f;
constexpr float LN2_ = 0.6931471805599453f;

constexpr int QTILE = 16;
constexpr int SCORE_STRIDE = 2052;    // f32 row stride, padded (+4) to stagger banks
constexpr int P_STRIDE = 2056;        // f16 row stride: 4112B rows -> 16B aligned, 4-bank stagger

// LDS layout (bytes)
constexpr int OFF_QT = 0;                                  // 16*64 f32   = 4096
constexpr int OFF_Q2 = OFF_QT + QTILE * D_ * 4;            // 16 f32 (pad to 128)
constexpr int OFF_SC = OFF_Q2 + 128;                       // 16*2052 f32 = 131328
constexpr int OFF_P  = OFF_SC + QTILE * SCORE_STRIDE * 4;  // 16*2056 f16 = 65792
constexpr int OFF_OB = OFF_P + QTILE * P_STRIDE * 2;       // 8*16*16 f32 = 8192
constexpr int SMEM_BYTES = OFF_OB + 8 * 16 * 16 * 4;       // ~210 KB < 320 KB WGP LDS

__global__ void __launch_bounds__(512, 1)
geo_attn_kernel(const float* __restrict__ qg, const float* __restrict__ kg,
                const float* __restrict__ vg, float* __restrict__ outg,
                float* __restrict__ attng)
{
    extern __shared__ char smem[];
    float*     qt = (float*)(smem + OFF_QT);
    float*     q2 = (float*)(smem + OFF_Q2);
    float*     sc = (float*)(smem + OFF_SC);
    _Float16*  pb = (_Float16*)(smem + OFF_P);
    float*     ob = (float*)(smem + OFF_OB);

    const int qb   = blockIdx.x;          // query tile 0..127
    const int b    = blockIdx.y;          // batch 0..7
    const int tid  = threadIdx.x;
    const int lane = tid & 31;
    const int wave = tid >> 5;
    const int m    = lane & 15;           // fragment row/col selector
    const int hi   = lane >> 4;           // lane half

    // ---- Phase 0: stage Q tile (16x64 f32 = 4KB) via async copy to LDS ------
    const float* qbase = qg + ((size_t)b * NQ + (size_t)qb * QTILE) * D_;
    {
        // 512 threads x 8B = 4KB; ASYNCcnt-tracked global->LDS DMA path
        const unsigned lds_dst = (unsigned)OFF_QT + (unsigned)tid * 8u;
        const unsigned long long ga = (unsigned long long)(const void*)(qbase + tid * 2);
        asm volatile("global_load_async_to_lds_b64 %0, %1, off"
                     :: "v"(lds_dst), "v"(ga) : "memory");
#if __has_builtin(__builtin_amdgcn_s_wait_asynccnt)
        __builtin_amdgcn_s_wait_asynccnt(0);
#else
        asm volatile("s_wait_asynccnt 0" ::: "memory");
#endif
    }
    __syncthreads();
    if (wave == 0 && lane < 16) {
        float s = 0.f;
        #pragma unroll
        for (int d = 0; d < D_; ++d) { float x = qt[lane * D_ + d]; s += x * x; }
        q2[lane] = s;
    }
    __syncthreads();

    // A-fragments for Q (16x32 f16 each; D=64 -> two fragments), same contiguous k order as B:
    // lane holds row M=lane&15; k = 16*hi + i.
    v16h aq0, aq1;
    #pragma unroll
    for (int i = 0; i < 16; ++i) {
        const int kk = (hi << 4) + i;
        aq0[i] = (_Float16)qt[m * D_ + kk];
        aq1[i] = (_Float16)qt[m * D_ + 32 + kk];
    }
    float q2r[8];
    #pragma unroll
    for (int r = 0; r < 8; ++r) q2r[r] = q2[r + (hi << 3)];

    // ---- Phase 1: QK^T + Poincare score per 16x16 tile -----------------------
    for (int t = 0; t < 8; ++t) {
        const int kt = wave * 8 + t;                 // 0..127
        const int n  = kt * QTILE + m;               // key row = B-fragment column
        const float* kb = kg + ((size_t)b * NK + n) * D_ + (hi << 4);
        if (t + 1 < 8) __builtin_prefetch(kb + (size_t)QTILE * D_, 0, 1);

        v16h bk0, bk1;
        float p2 = 0.f;
        #pragma unroll
        for (int i = 0; i < 16; ++i) {
            const float x0 = kb[i];
            const float x1 = kb[32 + i];
            bk0[i] = (_Float16)x0;
            bk1[i] = (_Float16)x1;
            p2 += x0 * x0 + x1 * x1;
        }
        const float y2 = p2 + __shfl_xor(p2, 16, 32);   // combine lane halves

        v8f c = {};
        c = __builtin_amdgcn_wmma_f32_16x16x32_f16(false, aq0, false, bk0,
                                                   (short)0, c, false, false);
        c = __builtin_amdgcn_wmma_f32_16x16x32_f16(false, aq1, false, bk1,
                                                   (short)0, c, false, false);

        const float omy = 1.f - y2;
        #pragma unroll
        for (int r = 0; r < 8; ++r) {
            const float x2 = q2r[r];
            const float xy = c[r];
            const float sq    = fmaxf(x2 + y2 - 2.f * xy, 0.f);
            const float denom = fmaxf((1.f - x2) * omy, EPS_);
            // fast-path transcendentals: v_rcp/v_sqrt/v_log instead of IEEE expansions
            const float arg = fmaxf(fmaf(2.f * sq, __builtin_amdgcn_rcpf(denom), 1.f),
                                    1.f + EPS_);
            const float dist = LN2_ * __builtin_amdgcn_logf(
                arg + __builtin_amdgcn_sqrtf(arg * arg - 1.f));   // arccosh
            sc[(r + (hi << 3)) * SCORE_STRIDE + kt * QTILE + m] = -dist * dist;
        }
    }
    __syncthreads();

    // ---- Phase 2: softmax, wave-per-row; stream attn weights to HBM ----------
    {
        const int row = wave;                        // 16 waves, 16 rows
        float mx = -3.402823e38f;
        for (int col = lane; col < NK; col += 32)
            mx = fmaxf(mx, sc[row * SCORE_STRIDE + col]);
        #pragma unroll
        for (int o = 16; o > 0; o >>= 1) mx = fmaxf(mx, __shfl_xor(mx, o, 32));

        float sum = 0.f;
        for (int col = lane; col < NK; col += 32) {
            const float e = __expf(sc[row * SCORE_STRIDE + col] - mx);
            sc[row * SCORE_STRIDE + col] = e;
            sum += e;
        }
        #pragma unroll
        for (int o = 16; o > 0; o >>= 1) sum += __shfl_xor(sum, o, 32);
        const float inv = __builtin_amdgcn_rcpf(sum);

        float* arow = attng + ((size_t)(b * NQ) + (size_t)qb * QTILE + row) * NK;
        for (int col = lane; col < NK; col += 32) {
            const float w = sc[row * SCORE_STRIDE + col] * inv;
            __builtin_nontemporal_store(w, arow + col);   // write-once 134MB stream
            pb[row * P_STRIDE + col] = (_Float16)w;
        }
    }
    __syncthreads();

    // ---- Phase 3: O = P(16x2048,f16) x V(2048x128) via WMMA ------------------
    {
        const int ntile = wave & 7;   // output column tile (8 x 16 = 128)
        const int khalf = wave >> 3;  // split K=2048 between wave pairs
        v8f c = {};
        const float* vb = vg + (size_t)b * NK * VDIM + ntile * 16 + m;

        for (int ks = khalf * 32; ks < khalf * 32 + 32; ++ks) {  // 32 K-steps of 32
            const int k0 = ks * 32;
            // A-fragment: two contiguous 8-half runs per lane -> 2x ds_load_b128
            const v8h alo = *(const v8h*)(pb + m * P_STRIDE + k0 + (hi << 3));
            const v8h ahi = *(const v8h*)(pb + m * P_STRIDE + k0 + 16 + (hi << 3));
            const v16h ap = __builtin_shufflevector(alo, ahi,
                0, 1, 2, 3, 4, 5, 6, 7, 8, 9, 10, 11, 12, 13, 14, 15);

            v16h bv;
            #pragma unroll
            for (int i = 0; i < 16; ++i) {
                const int vr = k0 + ((i >> 3) << 4) + (hi << 3) + (i & 7);
                bv[i] = (_Float16)vb[(size_t)vr * VDIM];
            }
            c = __builtin_amdgcn_wmma_f32_16x16x32_f16(false, ap, false, bv,
                                                       (short)0, c, false, false);
        }

        if (khalf == 1) {
            #pragma unroll
            for (int r = 0; r < 8; ++r)
                ob[ntile * 256 + (r + (hi << 3)) * 16 + m] = c[r];
        }
        __syncthreads();
        if (khalf == 0) {
            float* obase = outg + ((size_t)(b * NQ) + (size_t)qb * QTILE) * VDIM
                         + ntile * 16 + m;
            #pragma unroll
            for (int r = 0; r < 8; ++r) {
                const int mr = r + (hi << 3);
                obase[(size_t)mr * VDIM] = c[r] + ob[ntile * 256 + mr * 16 + m];
            }
        }
    }
}

extern "C" void kernel_launch(void* const* d_in, const int* in_sizes, int n_in,
                              void* d_out, int out_size, void* d_ws, size_t ws_size,
                              hipStream_t stream)
{
    const float* q = (const float*)d_in[0];
    const float* k = (const float*)d_in[1];
    const float* v = (const float*)d_in[2];
    float* out  = (float*)d_out;                       // (8,2048,128)
    float* attn = out + (size_t)B_ * NQ * VDIM;        // (8,2048,2048)

    dim3 grid(NQ / QTILE, B_);   // 128 x 8 workgroups
    dim3 block(512);             // 16 wave32s
    hipLaunchKernelGGL(geo_attn_kernel, grid, block, SMEM_BYTES, stream,
                       q, k, v, out, attn);
}
